// WMCA_6588479832344
// MI455X (gfx1250) — matrix operand order repaired and checked by reading, not compiled, not run on hardware
//
#include <hip/hip_runtime.h>

typedef _Float16 h16;
typedef __attribute__((ext_vector_type(16))) _Float16 v16h;
typedef __attribute__((ext_vector_type(8)))  _Float16 v8h;
typedef __attribute__((ext_vector_type(8)))  float     v8f;
typedef __attribute__((ext_vector_type(4)))  float     f32x4;

#define NROW   131072          // 2 * 1024 * 64 pixel-rows
#define NWIN   2048            // global windows (B * 1024)
#define CIN    192
#define HD     32
#define NH     6
#define ATT_SCALE 0.17677669529663687f   // 32^-0.5

// ---------------------------------------------------------------- helpers ---

static __device__ inline v8f wmma16(v16h a, v16h b, v8f c) {
    return __builtin_amdgcn_wmma_f32_16x16x32_f16(false, a, false, b, (short)0, c, false, false);
}

static __device__ inline v8h cvt8(const float* p) {
    f32x4 a = *(const f32x4*)p;
    f32x4 b = *(const f32x4*)(p + 4);
    v8h r;
    r[0]=(h16)a[0]; r[1]=(h16)a[1]; r[2]=(h16)a[2]; r[3]=(h16)a[3];
    r[4]=(h16)b[0]; r[5]=(h16)b[1]; r[6]=(h16)b[2]; r[7]=(h16)b[3];
    return r;
}

static __device__ inline v16h join8(v8h a, v8h b) {
    v16h r;
#pragma unroll
    for (int i = 0; i < 8; ++i) { r[i] = a[i]; r[i + 8] = b[i]; }
    return r;
}

// A operand 16x32 f16: lane holds row M = lane%16.
// elems 0..7  -> K = kb + hi*8 + e
// elems 8..15 -> K = kb + 16 + hi*8 + (e-8)      (hi = lane/16)
static __device__ inline v16h loadA_f32(const float* row, int kb, int hi) {
    return join8(cvt8(row + kb + hi * 8), cvt8(row + kb + 16 + hi * 8));
}
static __device__ inline v16h loadA_h(const h16* row, int kb, int hi) {
    return join8(*(const v8h*)(row + kb + hi * 8),
                 *(const v8h*)(row + kb + 16 + hi * 8));
}
// B operand 32x16 f16: lane holds col N = lane%16; elems 0..7 -> K = kb + hi*8 + e, elems 8..15 -> K = kb + 16 + hi*8 + (e-8)
static __device__ inline v16h loadB_f32(const float* row, int kb, int hi) {
    return join8(cvt8(row + kb + hi * 8), cvt8(row + kb + 16 + hi * 8));
}
static __device__ inline v16h loadB_h(const h16* row, int kb, int hi) {
    return join8(*(const v8h*)(row + kb + hi * 8),
                 *(const v8h*)(row + kb + 16 + hi * 8));
}

// row index R (0..NROW) -> float offset of the (rolled) source/dest pixel row.
// Folds the cyclic shift of +4 (roll by -4 on load side; same mapping works
// for the inverse-window + roll(+4) on the store side).
static __device__ inline size_t pixel_off(int R) {
    int b  = R >> 16;
    int wi = (R >> 6) & 1023;
    int p  = R & 63;
    int hh = (((wi >> 5) << 3) + (p >> 3) + 4) & 255;
    int ww = (((wi & 31) << 3) + (p & 7)  + 4) & 255;
    return (size_t)(((b << 8) | hh) << 8 | ww) * CIN;
}

// ---------------------------------------------------------------- kernels ---

// 0) gather relative-position bias table: tab[h][p][r], 6*64*64 f32
__global__ __launch_bounds__(256) void bias_gather(const float* __restrict__ rel,
                                                   float* __restrict__ tab) {
    int i = blockIdx.x * 256 + threadIdx.x;           // 24576 total
    int h = i >> 12, pr = i & 4095, p = pr >> 6, r = pr & 63;
    int pi = p >> 3, pj = p & 7, qi = r >> 3, qj = r & 7;
    tab[i] = rel[(h * 15 + (pi - qi + 7)) * 15 + (pj - qj + 7)];
}

// 1) Q / K projection: A = weight rows (M = out-channel), B = pixel rows.
//    dst layout: [h][R][c] halves (c contiguous).
__global__ __launch_bounds__(32) void proj_qk(const float* __restrict__ src,
                                              const float* __restrict__ w,
                                              const float* __restrict__ bias,
                                              h16* __restrict__ dst, int obase) {
    int lo = threadIdx.x & 15, hi = threadIdx.x >> 4;
    int ot = blockIdx.y;
    const float* wrow = w + (size_t)(obase + ot * 16 + lo) * CIN;
    v16h aF[6];
#pragma unroll
    for (int ks = 0; ks < 6; ++ks) aF[ks] = loadA_f32(wrow, ks * 32, hi);
    const float* bp = bias + obase + ot * 16 + 8 * hi;
    f32x4 bv0 = *(const f32x4*)bp;
    f32x4 bv1 = *(const f32x4*)(bp + 4);

    int orel = ot * 16 + 8 * hi;          // channel chunk (relative to obase)
    int hh = orel >> 5, cc = orel & 31;

#pragma unroll
    for (int rt = 0; rt < 4; ++rt) {
        int R = blockIdx.x * 64 + rt * 16 + lo;
        const float* srow = src + pixel_off(R);
        v8f acc = {};
#pragma unroll
        for (int ks = 0; ks < 6; ++ks)
            acc = wmma16(aF[ks], loadB_f32(srow, ks * 32, hi), acc);
        v8h o;
#pragma unroll
        for (int j = 0; j < 8; ++j)
            o[j] = (h16)(acc[j] + (j < 4 ? bv0[j] : bv1[j - 4]));
        *(v8h*)(dst + ((size_t)hh * NROW + R) * HD + cc) = o;
    }
}

// 2) V projection with transposed store: A = pixel rows, B = weight rows.
//    dst layout: Vt[h][win][c][p] halves (p contiguous).
__global__ __launch_bounds__(32) void proj_v(const float* __restrict__ src,
                                             const float* __restrict__ w,
                                             const float* __restrict__ bias,
                                             h16* __restrict__ vt) {
    int lo = threadIdx.x & 15, hi = threadIdx.x >> 4;
    int Rbase = blockIdx.x * 16;
    const float* srow = src + pixel_off(Rbase + lo);
    v16h aF[6];
#pragma unroll
    for (int ks = 0; ks < 6; ++ks) aF[ks] = loadA_f32(srow, ks * 32, hi);
    int Wg = Rbase >> 6;
    int pbase = (Rbase & 63) + 8 * hi;

#pragma unroll
    for (int ot = 0; ot < 12; ++ot) {
        const float* wrow = w + (size_t)(384 + ot * 16 + lo) * CIN;
        v8f acc = {};
#pragma unroll
        for (int ks = 0; ks < 6; ++ks)
            acc = wmma16(aF[ks], loadB_f32(wrow, ks * 32, hi), acc);
        float bb = bias[384 + ot * 16 + lo];
        int orel = ot * 16 + lo;
        int hh = orel >> 5, cc = orel & 31;
        v8h o;
#pragma unroll
        for (int j = 0; j < 8; ++j) o[j] = (h16)(acc[j] + bb);
        *(v8h*)(vt + (((size_t)hh * NWIN + Wg) * HD + cc) * 64 + pbase) = o;
    }
}

// 3) attention per (head, window): sim = qk^T*s + bias (+mask), softmax, @V
__global__ __launch_bounds__(32) void attn_k(const h16* __restrict__ Qb,
                                             const h16* __restrict__ Kb,
                                             const h16* __restrict__ Vt,
                                             const float* __restrict__ biasTab,
                                             h16* __restrict__ attn) {
    __shared__ __align__(16) h16 lds[16 * 64];
    int lo = threadIdx.x & 15, hi = threadIdx.x >> 4;
    int h  = blockIdx.x / NWIN;
    int Wg = blockIdx.x % NWIN;
    int wi = Wg & 1023;
    bool lastRow = ((wi >> 5) == 31), lastCol = ((wi & 31) == 31);

    const h16* kq = Qb + ((size_t)h * NROW + (size_t)Wg * 64) * HD;
    const h16* kk = Kb + ((size_t)h * NROW + (size_t)Wg * 64) * HD;
    const h16* vv = Vt + ((size_t)h * NWIN + Wg) * HD * 64;
    const float* bt = biasTab + h * 4096;
    h16* ob = attn + (size_t)Wg * 64 * CIN + h * HD;

    v16h kB[4];
#pragma unroll
    for (int t = 0; t < 4; ++t) kB[t] = loadB_h(kk + (t * 16 + lo) * HD, 0, hi);
    v16h vB[2][2];
#pragma unroll
    for (int ks = 0; ks < 2; ++ks)
#pragma unroll
        for (int ct = 0; ct < 2; ++ct)
            vB[ks][ct] = loadB_h(vv + (ct * 16 + lo) * 64, ks * 32, hi);

#pragma unroll
    for (int s = 0; s < 4; ++s) {
        v16h qA = loadA_h(kq + (s * 16 + lo) * HD, 0, hi);
        float sim[4][8];
#pragma unroll
        for (int t = 0; t < 4; ++t) {
            v8f acc = wmma16(qA, kB[t], v8f{});
#pragma unroll
            for (int j = 0; j < 8; ++j) sim[t][j] = acc[j];
        }
#pragma unroll
        for (int j = 0; j < 8; ++j) {
            int p  = s * 16 + j + 8 * hi;
            int pi = p >> 3, pj = p & 7;
            float m = -3.0e38f;
#pragma unroll
            for (int t = 0; t < 4; ++t) {
                int r = t * 16 + lo;
                float v = sim[t][j] * ATT_SCALE + bt[p * 64 + r];
                bool msk = (lastRow && ((pi < 4) != ((r >> 3) < 4))) ||
                           (lastCol && ((pj < 4) != ((r & 7) < 4)));
                v = msk ? -3.0e38f : v;
                sim[t][j] = v;
                m = fmaxf(m, v);
            }
#pragma unroll
            for (int off = 1; off < 16; off <<= 1) m = fmaxf(m, __shfl_xor(m, off, 32));
            float sum = 0.f;
#pragma unroll
            for (int t = 0; t < 4; ++t) { float e = __expf(sim[t][j] - m); sim[t][j] = e; sum += e; }
#pragma unroll
            for (int off = 1; off < 16; off <<= 1) sum += __shfl_xor(sum, off, 32);
            float inv = 1.0f / sum;
#pragma unroll
            for (int t = 0; t < 4; ++t)
                lds[(j + 8 * hi) * 64 + t * 16 + lo] = (h16)(sim[t][j] * inv);
        }
        asm volatile("s_wait_dscnt 0" ::: "memory");   // cross-lane LDS RAW
        v16h pA0 = loadA_h(lds + lo * 64, 0,  hi);
        v16h pA1 = loadA_h(lds + lo * 64, 32, hi);
        v8f o0 = wmma16(pA1, vB[1][0], wmma16(pA0, vB[0][0], v8f{}));
        v8f o1 = wmma16(pA1, vB[1][1], wmma16(pA0, vB[0][1], v8f{}));
#pragma unroll
        for (int j = 0; j < 8; ++j) {
            int p = s * 16 + j + 8 * hi;
            ob[(size_t)p * CIN + lo]      = (h16)o0[j];
            ob[(size_t)p * CIN + 16 + lo] = (h16)o1[j];
        }
    }
}

// 4) output projection + inverse window transform + roll(+4)
__global__ __launch_bounds__(32) void out_proj(const h16* __restrict__ attn,
                                               const float* __restrict__ w,
                                               const float* __restrict__ bias,
                                               float* __restrict__ out) {
    int lo = threadIdx.x & 15, hi = threadIdx.x >> 4;
    int ot = blockIdx.y;
    const float* wrow = w + (size_t)(ot * 16 + lo) * CIN;
    v16h aF[6];
#pragma unroll
    for (int ks = 0; ks < 6; ++ks) aF[ks] = loadA_f32(wrow, ks * 32, hi);
    const float* bp = bias + ot * 16 + 8 * hi;
    f32x4 bv0 = *(const f32x4*)bp;
    f32x4 bv1 = *(const f32x4*)(bp + 4);

#pragma unroll
    for (int rt = 0; rt < 4; ++rt) {
        int R = blockIdx.x * 64 + rt * 16 + lo;
        const h16* arow = attn + (size_t)R * CIN;
        v8f acc = {};
#pragma unroll
        for (int ks = 0; ks < 6; ++ks)
            acc = wmma16(aF[ks], loadB_h(arow, ks * 32, hi), acc);
        size_t off = pixel_off(R) + ot * 16 + 8 * hi;
        f32x4 o0, o1;
#pragma unroll
        for (int j = 0; j < 4; ++j) { o0[j] = acc[j] + bv0[j]; o1[j] = acc[j + 4] + bv1[j]; }
        *(f32x4*)(out + off)     = o0;
        *(f32x4*)(out + off + 4) = o1;
    }
}

// ----------------------------------------------------------------- launch ---

extern "C" void kernel_launch(void* const* d_in, const int* in_sizes, int n_in,
                              void* d_out, int out_size, void* d_ws, size_t ws_size,
                              hipStream_t stream) {
    (void)in_sizes; (void)n_in; (void)out_size; (void)ws_size;
    const float* x      = (const float*)d_in[0];
    const float* bsrc   = (const float*)d_in[1];
    const float* w_qkv  = (const float*)d_in[2];
    const float* b_qkv  = (const float*)d_in[3];
    const float* w_qkvb = (const float*)d_in[4];
    const float* b_qkvb = (const float*)d_in[5];
    const float* rel    = (const float*)d_in[6];
    const float* w_out  = (const float*)d_in[7];
    const float* b_out  = (const float*)d_in[8];
    float* out = (float*)d_out;

    char* ws = (char*)d_ws;
    float* biasTab = (float*)ws;                              // 98304 B
    h16* Qb = (h16*)(ws + (1u << 17));
    h16* Kb = Qb + (size_t)NH * NROW * HD;                    // 48 MiB each
    h16* Vt = Kb + (size_t)NH * NROW * HD;
    h16* At = Vt + (size_t)NH * NROW * HD;

    bias_gather<<<96, 256, 0, stream>>>(rel, biasTab);
    proj_qk<<<dim3(2048, 12), 32, 0, stream>>>(bsrc, w_qkvb, b_qkvb, Qb, 0);
    proj_qk<<<dim3(2048, 12), 32, 0, stream>>>(x,    w_qkv,  b_qkv,  Kb, 192);
    proj_v <<<8192, 32, 0, stream>>>(x, w_qkv, b_qkv, Vt);
    attn_k <<<NH * NWIN, 32, 0, stream>>>(Qb, Kb, Vt, biasTab, At);
    out_proj<<<dim3(2048, 12), 32, 0, stream>>>(At, w_out, b_out, out);
}
